// MemoryTransformerBlock_20933670601130
// MI455X (gfx1250) — matrix operand order repaired and checked by reading, not compiled, not run on hardware
//
#include <hip/hip_runtime.h>
#include <hip/hip_bf16.h>

// ---------------- problem constants ----------------
constexpr int Bc = 2, Sc = 1024, Dc = 1024, Hc = 16, Fc = 4096, Mc = 2048, DCc = 256;
constexpr int HDc = Dc / Hc;          // 64
constexpr int Nrows = Bc * Sc;        // 2048 token rows

typedef __attribute__((ext_vector_type(16))) __bf16 v16bf;
typedef __attribute__((ext_vector_type(8)))  float  v8f;
typedef __attribute__((ext_vector_type(4)))  int    v4i;
typedef __attribute__((address_space(1))) v4i v4i_glb;   // global int4
typedef __attribute__((address_space(3))) v4i v4i_lds;   // LDS int4

#if defined(__gfx1250__) && __has_builtin(__builtin_amdgcn_global_load_async_to_lds_b128)
#define USE_ASYNC_LDS 1
#else
#define USE_ASYNC_LDS 0
#endif

__device__ __forceinline__ v8f wmma_bf16(v16bf a, v16bf b, v8f c) {
  return __builtin_amdgcn_wmma_f32_16x16x32_bf16(false, a, false, b, (short)0, c, false, false);
}

union U4BF { uint4 u; __bf16 h[8]; };
union PairBF { unsigned u; __bf16 h[2]; };
union F4 { float4 f4; float f[4]; };

// build a 16-element bf16 fragment from two 16-byte chunks (LDS or global)
__device__ __forceinline__ v16bf frag_ld(const __bf16* p0, const __bf16* p1) {
  union { uint4 u[2]; v16bf v; } t;
  t.u[0] = *(const uint4*)p0;
  t.u[1] = *(const uint4*)p1;
  return t.v;
}

// ---------------- elementwise / norm kernels ----------------

__global__ void cvt_f32_bf16_kernel(const float* __restrict__ in, __bf16* __restrict__ out, size_t n) {
  size_t i = (size_t)blockIdx.x * blockDim.x + threadIdx.x;
  if (i < n) out[i] = (__bf16)in[i];
}

__global__ __launch_bounds__(256)
void rmsnorm_kernel(const float* __restrict__ x, const float* __restrict__ w,
                    __bf16* __restrict__ out, int D) {
  __shared__ float red[256];
  int r = blockIdx.x, tid = threadIdx.x;
  const float* xr = x + (size_t)r * D;
  float s = 0.f;
  for (int c = tid; c < D; c += 256) { float v = xr[c]; s += v * v; }
  red[tid] = s; __syncthreads();
  for (int off = 128; off; off >>= 1) { if (tid < off) red[tid] += red[tid + off]; __syncthreads(); }
  float scale = rsqrtf(red[0] / (float)D + 1e-6f);
  for (int c = tid; c < D; c += 256) out[(size_t)r * D + c] = (__bf16)(xr[c] * scale * w[c]);
}

__global__ void rope_kernel(const float* __restrict__ x, const float* __restrict__ cosb,
                            const float* __restrict__ sinb, __bf16* __restrict__ out) {
  size_t i = (size_t)blockIdx.x * 256 + threadIdx.x;
  size_t total = (size_t)Bc * Sc * Dc;
  if (i >= total) return;
  int d = (int)(i % HDc);
  int s = (int)((i / Dc) % Sc);
  float c = cosb[s * HDc + d], sn = sinb[s * HDc + d];
  float xv = x[i];
  float xr = (d < HDc / 2) ? -x[i + HDc / 2] : x[i - HDc / 2];
  out[i] = (__bf16)(xv * c + xr * sn);
}

__global__ __launch_bounds__(256)
void strength_kernel(const float* __restrict__ h, const float* __restrict__ wst,
                     float* __restrict__ out) {
  __shared__ float red[256];
  int r = blockIdx.x, tid = threadIdx.x;
  const float* hr = h + (size_t)r * Dc;
  float s = 0.f;
  for (int c = tid; c < Dc; c += 256) s += hr[c] * wst[c];
  red[tid] = s; __syncthreads();
  for (int off = 128; off; off >>= 1) { if (tid < off) red[tid] += red[tid + off]; __syncthreads(); }
  if (tid == 0) out[r] = 1.f / (1.f + __expf(-red[0]));
}

__global__ void memupd_kernel(const float* __restrict__ upd, const float* __restrict__ strength,
                              float* __restrict__ dst, const int* __restrict__ pos, int cols) {
  size_t i = (size_t)blockIdx.x * 256 + threadIdx.x;
  size_t total = (size_t)Bc * Sc * cols;
  if (i >= total) return;
  int c = (int)(i % cols);
  int s = (int)((i / cols) % Sc);
  int b = (int)(i / ((size_t)cols * Sc));
  int row = (pos[0] + s) % Mc;
  float st = strength[b * Sc + s];
  size_t di = ((size_t)b * Mc + row) * cols + c;
  dst[di] = st * upd[i] + (1.f - st) * dst[di];
}

__global__ __launch_bounds__(256)
void softmax_kernel(const float* __restrict__ sc, __bf16* __restrict__ out, int cols) {
  __shared__ float red[256];
  int r = blockIdx.x, tid = threadIdx.x;
  const float* row = sc + (size_t)r * cols;
  float mx = -1e30f;
  for (int c = tid; c < cols; c += 256) mx = fmaxf(mx, row[c]);
  red[tid] = mx; __syncthreads();
  for (int off = 128; off; off >>= 1) { if (tid < off) red[tid] = fmaxf(red[tid], red[tid + off]); __syncthreads(); }
  mx = red[0]; __syncthreads();
  float s = 0.f;
  for (int c = tid; c < cols; c += 256) s += __expf(row[c] - mx);
  red[tid] = s; __syncthreads();
  for (int off = 128; off; off >>= 1) { if (tid < off) red[tid] += red[tid + off]; __syncthreads(); }
  float inv = 1.f / red[0];
  for (int c = tid; c < cols; c += 256) out[(size_t)r * cols + c] = (__bf16)(__expf(row[c] - mx) * inv);
}

__global__ void gatecomb_kernel(const float* __restrict__ h1, const float* __restrict__ gl,
                                const float* __restrict__ proj, float* __restrict__ h2) {
  size_t i = (size_t)blockIdx.x * 256 + threadIdx.x;
  size_t total = (size_t)Nrows * Dc;
  if (i >= total) return;
  float g = 1.f / (1.f + __expf(-gl[i]));
  h2[i] = h1[i] + g * proj[i];
}

__global__ void silu_mul_kernel(const __bf16* __restrict__ a1, const __bf16* __restrict__ a3,
                                __bf16* __restrict__ out, size_t n) {
  size_t i = (size_t)blockIdx.x * 256 + threadIdx.x;
  if (i >= n) return;
  float x = (float)a1[i];
  out[i] = (__bf16)((x / (1.f + __expf(-x))) * (float)a3[i]);
}

// ---------------- generic WMMA GEMM  C[M,N] = alpha * A[M,K] x B[K,N] (+res) ----------------
// flags: bit0 = B is f32 (convert during LDS staging), bit1 = B stored transposed (N x K rows)
// B tile is held in LDS TRANSPOSED (BsT[n][k]) so every WMMA B-fragment is two ds_load_b128.

#define BM 64
#define BN 64
#define BKk 32

__global__ __launch_bounds__(128)
void gemm_wmma_kernel(const __bf16* __restrict__ A, int lda,
                      const void* __restrict__ Bv, int ldb, int flags,
                      const float* __restrict__ res, float* __restrict__ outf,
                      __bf16* __restrict__ outbf, int ldc, int K, float alpha) {
  __shared__ __bf16 As[BM][BKk];     // A tile, row-major
  __shared__ __bf16 BsT[BN][BKk];    // B tile, transposed: BsT[n][k]
  int tid = threadIdx.x;
  int wave = tid >> 5, lane = tid & 31;
  int m0 = blockIdx.y * BM, n0 = blockIdx.x * BN;
  int mloc = lane & 15, khalf = lane >> 4, koff = khalf * 8;
  const __bf16* Bb = (const __bf16*)Bv;
  const float*  Bf = (const float*)Bv;
  v8f acc[4] = {v8f{}, v8f{}, v8f{}, v8f{}};

  for (int k0 = 0; k0 < K; k0 += BKk) {
    { // ---- stage A: each thread copies 16 contiguous bf16 (2 x b128) ----
      int mm = tid >> 1, kk0 = (tid & 1) * 16;
      const __bf16* asrc = A + (size_t)(m0 + mm) * lda + k0 + kk0;
#if USE_ASYNC_LDS
      __builtin_amdgcn_global_load_async_to_lds_b128(
          (v4i_glb*)asrc, (v4i_lds*)&As[mm][kk0], 0, 0);
      __builtin_amdgcn_global_load_async_to_lds_b128(
          (v4i_glb*)(asrc + 8), (v4i_lds*)&As[mm][kk0 + 8], 0, 0);
#else
      ((uint4*)&As[mm][kk0])[0] = ((const uint4*)asrc)[0];
      ((uint4*)&As[mm][kk0])[1] = ((const uint4*)asrc)[1];
      if (k0 + BKk < K)
        __builtin_prefetch((const void*)(asrc + BKk), 0, 1);
#endif
    }
    // ---- stage B into transposed LDS tile ----
    if (!(flags & 1)) {            // bf16 row-major: read 8-wide from two k rows, pack b32 pairs
      int nn0 = (tid & 7) * 8, kk = (tid >> 3) * 2;
      U4BF ra, rb;
      ra.u = *(const uint4*)(Bb + (size_t)(k0 + kk) * ldb + n0 + nn0);
      rb.u = *(const uint4*)(Bb + (size_t)(k0 + kk + 1) * ldb + n0 + nn0);
      for (int j = 0; j < 8; j++) {
        PairBF p; p.h[0] = ra.h[j]; p.h[1] = rb.h[j];
        *(unsigned*)&BsT[nn0 + j][kk] = p.u;
      }
    } else if (!(flags & 2)) {     // f32 row-major
      int nn0 = (tid & 7) * 8, kk = (tid >> 3) * 2;
      const float* sa = Bf + (size_t)(k0 + kk) * ldb + n0 + nn0;
      const float* sb = sa + ldb;
      for (int j = 0; j < 8; j++) {
        PairBF p; p.h[0] = (__bf16)sa[j]; p.h[1] = (__bf16)sb[j];
        *(unsigned*)&BsT[nn0 + j][kk] = p.u;
      }
    } else {                       // f32 stored [N][K]: direct copy into BsT row
      int nn = tid & 63, kk0b = (tid >> 6) * 16;
      const float* src = Bf + (size_t)(n0 + nn) * ldb + k0 + kk0b;
      U4BF w0, w1;
      for (int j = 0; j < 8; j++) { w0.h[j] = (__bf16)src[j]; w1.h[j] = (__bf16)src[8 + j]; }
      *(uint4*)&BsT[nn][kk0b] = w0.u;
      *(uint4*)&BsT[nn][kk0b + 8] = w1.u;
    }
#if USE_ASYNC_LDS
    asm volatile("s_wait_asynccnt 0x0" ::: "memory");
#endif
    __syncthreads();

    // ---- compute: A-frag + 4 B-frags, all 2 x ds_load_b128 each ----
    const __bf16* arow = &As[wave * 16 + mloc][0];
    v16bf a = frag_ld(arow + khalf * 16, arow + khalf * 16 + 8);
    for (int nt = 0; nt < 4; nt++) {
      const __bf16* brow = &BsT[nt * 16 + mloc][0] + khalf * 16;
      v16bf bfr = frag_ld(brow, brow + 8);
      acc[nt] = wmma_bf16(a, bfr, acc[nt]);
    }
    __syncthreads();
  }

  for (int nt = 0; nt < 4; nt++) {
    for (int r = 0; r < 8; r++) {
      size_t row = (size_t)m0 + wave * 16 + r + khalf * 8;
      size_t col = (size_t)n0 + nt * 16 + mloc;
      size_t idx = row * (size_t)ldc + col;
      float v = acc[nt][r] * alpha;
      if (res)   v += res[idx];
      if (outf)  outf[idx] = v;
      if (outbf) outbf[idx] = (__bf16)v;
    }
  }
}

// ---------------- causal attention: one wave per (b, h, 16-query tile) ----------------

__global__ __launch_bounds__(32)
void attn_kernel(const __bf16* __restrict__ qbf, const __bf16* __restrict__ kbf,
                 const __bf16* __restrict__ vbf, __bf16* __restrict__ out) {
  __shared__ float sc[16][Sc];       // 64 KB score rows for this query tile
  __shared__ __bf16 VsT[64][BKk];    // transposed V chunk: VsT[headcol][key-in-chunk]
  int qt = blockIdx.x, h = blockIdx.y, b = blockIdx.z;
  int lane = threadIdx.x;
  int mloc = lane & 15, khalf = lane >> 4, koff = khalf * 8;

  // load Q fragments (16x64 split into two 16x32 A-frags), vectorized b128 loads
  const __bf16* qrow = qbf + ((size_t)(b * Sc + qt * 16 + mloc) * Dc + h * HDc);
  v16bf qa0 = frag_ld(qrow + khalf * 16, qrow + khalf * 16 + 8);
  v16bf qa1 = frag_ld(qrow + 32 + khalf * 16, qrow + 32 + khalf * 16 + 8);

  int nkt = qt + 1;  // causal: only key tiles <= query tile
  for (int kb = 0; kb < nkt; kb++) {
    const __bf16* krow = kbf + ((size_t)(b * Sc + kb * 16 + mloc) * Dc + h * HDc);
    v16bf b0 = frag_ld(krow + khalf * 16, krow + khalf * 16 + 8);
    v16bf b1 = frag_ld(krow + 32 + khalf * 16, krow + 32 + khalf * 16 + 8);
    v8f c = v8f{};
    c = wmma_bf16(qa0, b0, c);
    c = wmma_bf16(qa1, b1, c);
    for (int r = 0; r < 8; r++) {
      int ml = r + khalf * 8;
      int mg = qt * 16 + ml;
      int ng = kb * 16 + mloc;
      sc[ml][kb * 16 + mloc] = (mg >= ng) ? c[r] * 0.125f : -1e30f;
    }
  }
  __syncthreads();

  int L = nkt * 16;
  if (lane < 16) {  // per-row softmax over the valid prefix
    float mx = -1e30f;
    for (int j = 0; j < L; j++) mx = fmaxf(mx, sc[lane][j]);
    float sum = 0.f;
    for (int j = 0; j < L; j++) { float e = __expf(sc[lane][j] - mx); sc[lane][j] = e; sum += e; }
    float inv = 1.f / sum;
    for (int j = 0; j < L; j++) sc[lane][j] *= inv;
  }
  __syncthreads();
  if (nkt & 1) {  // zero-pad an extra 16-key tile so P chunks are 32 wide
    for (int i = lane; i < 16 * 16; i += 32) sc[i >> 4][L + (i & 15)] = 0.f;
    __syncthreads();
  }

  int nch = (nkt + 1) >> 1;
  v8f acc[4] = {v8f{}, v8f{}, v8f{}, v8f{}};
  for (int ch = 0; ch < nch; ch++) {
    { // stage transposed V chunk: each lane copies its own V row (8 x b128 global loads)
      const __bf16* vrow = vbf + (size_t)(b * Sc + ch * 32 + lane) * Dc + h * HDc;
      U4BF rv[8];
      for (int c4 = 0; c4 < 8; c4++) rv[c4].u = ((const uint4*)vrow)[c4];
      for (int j = 0; j < 64; j++) VsT[j][lane] = rv[j >> 3].h[j & 7];
    }
    __syncthreads();

    // P fragment from LDS score rows (float4 loads + convert)
    F4 p0, p1, p2, p3;
    p0.f4 = *(const float4*)&sc[mloc][ch * 32 + koff];
    p1.f4 = *(const float4*)&sc[mloc][ch * 32 + koff + 4];
    p2.f4 = *(const float4*)&sc[mloc][ch * 32 + 16 + koff];
    p3.f4 = *(const float4*)&sc[mloc][ch * 32 + 16 + koff + 4];
    v16bf pa;
    for (int j = 0; j < 4; j++) {
      pa[j]      = (__bf16)p0.f[j];
      pa[4 + j]  = (__bf16)p1.f[j];
      pa[8 + j]  = (__bf16)p2.f[j];
      pa[12 + j] = (__bf16)p3.f[j];
    }
    for (int nt = 0; nt < 4; nt++) {
      const __bf16* brow = &VsT[nt * 16 + mloc][0] + koff;
      v16bf vb = frag_ld(brow, brow + 16);
      acc[nt] = wmma_bf16(pa, vb, acc[nt]);
    }
    __syncthreads();
  }

  for (int nt = 0; nt < 4; nt++)
    for (int r = 0; r < 8; r++)
      out[(size_t)(b * Sc + qt * 16 + r + khalf * 8) * Dc + h * HDc + nt * 16 + mloc] =
          (__bf16)acc[nt][r];
}

// ---------------- host orchestration ----------------

extern "C" void kernel_launch(void* const* d_in, const int* in_sizes, int n_in,
                              void* d_out, int out_size, void* d_ws, size_t ws_size,
                              hipStream_t stream) {
  (void)in_sizes; (void)n_in; (void)out_size; (void)ws_size;
  const float* hidden     = (const float*)d_in[0];
  const float* mem_keys   = (const float*)d_in[1];
  const float* mem_values = (const float*)d_in[2];
  const float* cosb       = (const float*)d_in[3];
  const float* sinb       = (const float*)d_in[4];
  const float* norm1_w    = (const float*)d_in[5];
  const float* norm2_w    = (const float*)d_in[6];
  const float* norm_mem_w = (const float*)d_in[7];
  const float* Wq   = (const float*)d_in[8];
  const float* Wk   = (const float*)d_in[9];
  const float* Wv   = (const float*)d_in[10];
  const float* Wo   = (const float*)d_in[11];
  const float* Wcnt = (const float*)d_in[12];
  const float* Wcmp = (const float*)d_in[13];
  const float* Wstr = (const float*)d_in[14];
  const float* Wqry = (const float*)d_in[15];
  const float* Wgat = (const float*)d_in[16];
  const float* Wprj = (const float*)d_in[17];
  const float* W1   = (const float*)d_in[18];
  const float* W3   = (const float*)d_in[19];
  const float* W2   = (const float*)d_in[20];
  const int*   pos  = (const int*)d_in[21];

  float* out_h    = (float*)d_out;
  float* out_keys = out_h + (size_t)Nrows * Dc;
  float* out_vals = out_keys + (size_t)Bc * Mc * DCc;

  char* wsp = (char*)d_ws;
  auto alloc = [&](size_t bytes) -> void* {
    void* p = (void*)wsp;
    wsp += (bytes + 255) & ~(size_t)255;
    return p;
  };

  // bf16 weight copies
  __bf16* Wq_b   = (__bf16*)alloc((size_t)Dc * Dc * 2);
  __bf16* Wk_b   = (__bf16*)alloc((size_t)Dc * Dc * 2);
  __bf16* Wv_b   = (__bf16*)alloc((size_t)Dc * Dc * 2);
  __bf16* Wo_b   = (__bf16*)alloc((size_t)Dc * Dc * 2);
  __bf16* Wcnt_b = (__bf16*)alloc((size_t)Dc * Dc * 2);
  __bf16* Wprj_b = (__bf16*)alloc((size_t)Dc * Dc * 2);
  __bf16* Wcmp_b = (__bf16*)alloc((size_t)Dc * DCc * 2);
  __bf16* Wqry_b = (__bf16*)alloc((size_t)Dc * DCc * 2);
  __bf16* Wgat_b = (__bf16*)alloc((size_t)2 * Dc * Dc * 2);
  __bf16* W1_b   = (__bf16*)alloc((size_t)Dc * Fc * 2);
  __bf16* W3_b   = (__bf16*)alloc((size_t)Dc * Fc * 2);
  __bf16* W2_b   = (__bf16*)alloc((size_t)Fc * Dc * 2);
  // activations
  __bf16* nbf    = (__bf16*)alloc((size_t)Nrows * Dc * 2);
  float*  qf     = (float*) alloc((size_t)Nrows * Dc * 4);
  float*  kf     = (float*) alloc((size_t)Nrows * Dc * 4);
  __bf16* qbf    = (__bf16*)alloc((size_t)Nrows * Dc * 2);
  __bf16* kbf    = (__bf16*)alloc((size_t)Nrows * Dc * 2);
  __bf16* vbf    = (__bf16*)alloc((size_t)Nrows * Dc * 2);
  __bf16* attnbf = (__bf16*)alloc((size_t)Nrows * Dc * 2);
  float*  h1f    = (float*) alloc((size_t)Nrows * Dc * 4);
  __bf16* h1bf   = (__bf16*)alloc((size_t)Nrows * Dc * 2);
  float*  cntf   = (float*) alloc((size_t)Nrows * Dc * 4);
  float*  cmpf   = (float*) alloc((size_t)Nrows * DCc * 4);
  float*  strf   = (float*) alloc((size_t)Nrows * 4);
  __bf16* nmbf   = (__bf16*)alloc((size_t)Nrows * Dc * 2);
  __bf16* qmbf   = (__bf16*)alloc((size_t)Nrows * DCc * 2);
  float*  mscf   = (float*) alloc((size_t)Nrows * Mc * 4);
  __bf16* prbbf  = (__bf16*)alloc((size_t)Nrows * Mc * 2);
  float*  retrf  = (float*) alloc((size_t)Nrows * Dc * 4);
  __bf16* retrbf = (__bf16*)alloc((size_t)Nrows * Dc * 2);
  float*  gatef  = (float*) alloc((size_t)Nrows * Dc * 4);
  float*  projf  = (float*) alloc((size_t)Nrows * Dc * 4);
  float*  h2f    = (float*) alloc((size_t)Nrows * Dc * 4);
  __bf16* n2bf   = (__bf16*)alloc((size_t)Nrows * Dc * 2);
  __bf16* a1bf   = (__bf16*)alloc((size_t)Nrows * Fc * 2);
  __bf16* a3bf   = (__bf16*)alloc((size_t)Nrows * Fc * 2);
  __bf16* mbf    = (__bf16*)alloc((size_t)Nrows * Fc * 2);

  auto cvt = [&](const float* src, __bf16* dst, size_t n) {
    cvt_f32_bf16_kernel<<<(int)((n + 255) / 256), 256, 0, stream>>>(src, dst, n);
  };
  auto gemm = [&](const __bf16* A, int lda, const void* Bp, int ldb, int flags,
                  const float* res, float* outf, __bf16* outbf, int ldc,
                  int Mr, int Nc, int K, float alpha) {
    dim3 g(Nc / 64, Mr / 64);
    gemm_wmma_kernel<<<g, 128, 0, stream>>>(A, lda, Bp, ldb, flags, res, outf, outbf, ldc, K, alpha);
  };

  // weights -> bf16
  cvt(Wq, Wq_b, (size_t)Dc * Dc);     cvt(Wk, Wk_b, (size_t)Dc * Dc);
  cvt(Wv, Wv_b, (size_t)Dc * Dc);     cvt(Wo, Wo_b, (size_t)Dc * Dc);
  cvt(Wcnt, Wcnt_b, (size_t)Dc * Dc); cvt(Wprj, Wprj_b, (size_t)Dc * Dc);
  cvt(Wcmp, Wcmp_b, (size_t)Dc * DCc); cvt(Wqry, Wqry_b, (size_t)Dc * DCc);
  cvt(Wgat, Wgat_b, (size_t)2 * Dc * Dc);
  cvt(W1, W1_b, (size_t)Dc * Fc);  cvt(W3, W3_b, (size_t)Dc * Fc);
  cvt(W2, W2_b, (size_t)Fc * Dc);

  // n = rmsnorm(h, norm1_w)
  rmsnorm_kernel<<<Nrows, 256, 0, stream>>>(hidden, norm1_w, nbf, Dc);
  // q,k (f32 for RoPE), v directly bf16
  gemm(nbf, Dc, Wq_b, Dc, 0, nullptr, qf, nullptr, Dc, Nrows, Dc, Dc, 1.f);
  gemm(nbf, Dc, Wk_b, Dc, 0, nullptr, kf, nullptr, Dc, Nrows, Dc, Dc, 1.f);
  gemm(nbf, Dc, Wv_b, Dc, 0, nullptr, nullptr, vbf, Dc, Nrows, Dc, Dc, 1.f);
  {
    int nb = (int)(((size_t)Nrows * Dc + 255) / 256);
    rope_kernel<<<nb, 256, 0, stream>>>(qf, cosb, sinb, qbf);
    rope_kernel<<<nb, 256, 0, stream>>>(kf, cosb, sinb, kbf);
  }
  // causal attention
  attn_kernel<<<dim3(Sc / 16, Hc, Bc), 32, 0, stream>>>(qbf, kbf, vbf, attnbf);
  // h1 = hidden + attn @ Wo
  gemm(attnbf, Dc, Wo_b, Dc, 0, hidden, h1f, h1bf, Dc, Nrows, Dc, Dc, 1.f);
  // memory write inputs
  gemm(h1bf, Dc, Wcnt_b, Dc, 0, nullptr, cntf, nullptr, Dc, Nrows, Dc, Dc, 1.f);
  gemm(h1bf, Dc, Wcmp_b, DCc, 0, nullptr, cmpf, nullptr, DCc, Nrows, DCc, Dc, 1.f);
  strength_kernel<<<Nrows, 256, 0, stream>>>(h1f, Wstr, strf);
  // new_keys / new_values : copy old then blend updated rows
  (void)hipMemcpyAsync(out_keys, mem_keys, (size_t)Bc * Mc * DCc * 4, hipMemcpyDeviceToDevice, stream);
  (void)hipMemcpyAsync(out_vals, mem_values, (size_t)Bc * Mc * Dc * 4, hipMemcpyDeviceToDevice, stream);
  memupd_kernel<<<(int)(((size_t)Bc * Sc * DCc + 255) / 256), 256, 0, stream>>>(cmpf, strf, out_keys, pos, DCc);
  memupd_kernel<<<(int)(((size_t)Bc * Sc * Dc + 255) / 256), 256, 0, stream>>>(cntf, strf, out_vals, pos, Dc);
  // memory attention
  rmsnorm_kernel<<<Nrows, 256, 0, stream>>>(h1f, norm_mem_w, nmbf, Dc);
  gemm(nmbf, Dc, Wqry_b, DCc, 0, nullptr, nullptr, qmbf, DCc, Nrows, DCc, Dc, 1.f);
  for (int b = 0; b < Bc; b++) {
    gemm(qmbf + (size_t)b * Sc * DCc, DCc,
         out_keys + (size_t)b * Mc * DCc, DCc, /*f32|trans*/ 3,
         nullptr, mscf + (size_t)b * Sc * Mc, nullptr, Mc, Sc, Mc, DCc, 0.0625f);
  }
  softmax_kernel<<<Nrows, 256, 0, stream>>>(mscf, prbbf, Mc);
  for (int b = 0; b < Bc; b++) {
    gemm(prbbf + (size_t)b * Sc * Mc, Mc,
         out_vals + (size_t)b * Mc * Dc, Dc, /*f32*/ 1,
         nullptr, retrf + (size_t)b * Sc * Dc, retrbf + (size_t)b * Sc * Dc, Dc, Sc, Dc, Mc, 1.f);
  }
  // gate = sigmoid([h1, retrieved] @ W_gate); two accumulated GEMMs
  gemm(h1bf, Dc, Wgat_b, Dc, 0, nullptr, gatef, nullptr, Dc, Nrows, Dc, Dc, 1.f);
  gemm(retrbf, Dc, Wgat_b + (size_t)Dc * Dc, Dc, 0, gatef, gatef, nullptr, Dc, Nrows, Dc, Dc, 1.f);
  gemm(retrbf, Dc, Wprj_b, Dc, 0, nullptr, projf, nullptr, Dc, Nrows, Dc, Dc, 1.f);
  gatecomb_kernel<<<(int)(((size_t)Nrows * Dc + 255) / 256), 256, 0, stream>>>(h1f, gatef, projf, h2f);
  // FFN: h_out = h2 + (silu(n2@W1) * (n2@W3)) @ W2
  rmsnorm_kernel<<<Nrows, 256, 0, stream>>>(h2f, norm2_w, n2bf, Dc);
  gemm(n2bf, Dc, W1_b, Fc, 0, nullptr, nullptr, a1bf, Fc, Nrows, Fc, Dc, 1.f);
  gemm(n2bf, Dc, W3_b, Fc, 0, nullptr, nullptr, a3bf, Fc, Nrows, Fc, Dc, 1.f);
  silu_mul_kernel<<<(int)(((size_t)Nrows * Fc + 255) / 256), 256, 0, stream>>>(a1bf, a3bf, mbf, (size_t)Nrows * Fc);
  gemm(mbf, Fc, W2_b, Dc, 0, h2f, out_h, nullptr, Dc, Nrows, Dc, Fc, 1.f);
}
